// WindowAttention_17428977287303
// MI455X (gfx1250) — compile-verified, matrix operand order checked by reading, not run on hardware
//
#include <hip/hip_runtime.h>
#include <hip/hip_bf16.h>

// ---------------- problem constants ----------------
#define DIMC   512
#define HEADS  16
#define HD     32
#define SWIN   49
#define SPAD   64
#define NWIN   64
#define BWIN   2048                 // total windows
#define MROWS  (BWIN * SWIN)       // 100352 = 784 * 128
#define NQKV   (3 * DIMC)          // 1536
#define TILE_ELEMS (SPAD * HD)     // 2048 bf16 per (b,h) tile

// workspace layout (bytes)
#define ADD_BYTES   ((size_t)NWIN * HEADS * SPAD * SPAD * 4)          // 16,777,216
#define QKV_OFF     ADD_BYTES
#define QKV_ELEMS   ((size_t)3 * BWIN * HEADS * TILE_ELEMS)           // 201,326,592 bf16
#define ATT_OFF     (QKV_OFF + QKV_ELEMS * 2)                         // 419,430,400
// attn-out: MROWS*512 bf16 = 102,760,448 bytes ; total ws ~498 MiB

typedef __bf16 v16bf __attribute__((ext_vector_type(16)));
typedef __bf16 v8bf  __attribute__((ext_vector_type(8)));
typedef float  v8f   __attribute__((ext_vector_type(8)));

static __device__ __forceinline__ v8f wmma_bf16(v16bf a, v16bf b, v8f c) {
  return __builtin_amdgcn_wmma_f32_16x16x32_bf16(false, a, false, b, (short)0, c,
                                                 false, false);
}

// A-frag (16x32, M from lane%16) and B-frag (32x16, N from lane%16) both read
// base[row][kb..kb+7] and [kb+16..kb+23] with kb = (lane<16 ? 0 : 8) (+kstep).
static __device__ __forceinline__ v16bf load_frag(const __bf16* p, int row,
                                                  int stride, int kb) {
  const __bf16* q = p + row * stride + kb;
  v8bf lo = *(const v8bf*)q;
  v8bf hi = *(const v8bf*)(q + 16);
  return __builtin_shufflevector(lo, hi, 0, 1, 2, 3, 4, 5, 6, 7,
                                 8, 9, 10, 11, 12, 13, 14, 15);
}

// ---------------- kernel 0a: bias+mask table, padded with -1e30 ----------------
__global__ __launch_bounds__(256) void build_add(const float* __restrict__ bias_table,
                                                 const int* __restrict__ rel_idx,
                                                 const float* __restrict__ mask,
                                                 float* __restrict__ addtab) {
  int idx = blockIdx.x * 256 + threadIdx.x;          // (w,h,m,n) n fastest
  int n = idx & 63;
  int m = (idx >> 6) & 63;
  int h = (idx >> 12) & 15;
  int w = idx >> 16;
  float v = -1e30f;
  if (m < SWIN && n < SWIN) {
    int ri = rel_idx[m * SWIN + n];
    v = bias_table[ri * HEADS + h] + mask[(w * SWIN + m) * SWIN + n];
  }
  addtab[idx] = v;
}

// ---------------- kernel 0b: zero the pad rows (s=49..63) of all qkv tiles ----
__global__ __launch_bounds__(256) void zero_pads(__bf16* __restrict__ qkv) {
  int idx = blockIdx.x * 256 + threadIdx.x;          // 98304 tiles * 60 chunks
  int tile = idx / 60;
  int chunk = idx - tile * 60;
  v8bf z = {};
  *(v8bf*)(qkv + (size_t)tile * TILE_ELEMS + SWIN * HD + chunk * 8) = z;
}

// ---------------- kernel 1: qkv = x @ W + b, scatter to padded bf16 tiles -----
// grid (784, 12), block 256 (8 waves); block tile 128x128, K-tile 64
__global__ __launch_bounds__(256) void qkv_gemm(const float* __restrict__ x,
                                                const float* __restrict__ w,
                                                const float* __restrict__ bias,
                                                __bf16* __restrict__ qkv) {
  __shared__ __bf16 As[128 * 72];   // row-major [m][k]
  __shared__ __bf16 Bs[128 * 72];   // transposed [n][k]
  const int tid = threadIdx.x;
  const int lane = tid & 31, wave = tid >> 5;
  const int l16 = lane & 15;
  const int kb0 = (lane >> 4) * 8;
  const int half8 = (lane >> 4) * 8;
  const int bm = blockIdx.x * 128;
  const int bn = blockIdx.y * 128;
  const int wm = (wave >> 1) * 32, wn = (wave & 1) * 64;

  v8f acc[2][4];
#pragma unroll
  for (int i = 0; i < 2; ++i)
#pragma unroll
    for (int j = 0; j < 4; ++j) acc[i][j] = (v8f){};

  for (int k0 = 0; k0 < DIMC; k0 += 64) {
    __syncthreads();
    {
      int r = tid >> 4;                 // A tile: 128 x 64 f32 -> bf16
      int c = (tid & 15) * 4;
#pragma unroll
      for (int p = 0; p < 8; ++p, r += 16) {
        float4 f = *(const float4*)(x + (size_t)(bm + r) * DIMC + k0 + c);
        __bf16* d = &As[r * 72 + c];
        d[0] = (__bf16)f.x; d[1] = (__bf16)f.y; d[2] = (__bf16)f.z; d[3] = (__bf16)f.w;
      }
      int kr = tid >> 5;                // B tile: 64 x 128 f32 -> transposed bf16
      int nc = (tid & 31) * 4;
#pragma unroll
      for (int p = 0; p < 8; ++p, kr += 8) {
        float4 f = *(const float4*)(w + (size_t)(k0 + kr) * NQKV + bn + nc);
        Bs[(nc + 0) * 72 + kr] = (__bf16)f.x;
        Bs[(nc + 1) * 72 + kr] = (__bf16)f.y;
        Bs[(nc + 2) * 72 + kr] = (__bf16)f.z;
        Bs[(nc + 3) * 72 + kr] = (__bf16)f.w;
      }
    }
    __syncthreads();
#pragma unroll
    for (int ks = 0; ks < 64; ks += 32) {
      v16bf a[2], b[4];
#pragma unroll
      for (int i = 0; i < 2; ++i) a[i] = load_frag(As, wm + i * 16 + l16, 72, ks + kb0);
#pragma unroll
      for (int j = 0; j < 4; ++j) b[j] = load_frag(Bs, wn + j * 16 + l16, 72, ks + kb0);
#pragma unroll
      for (int i = 0; i < 2; ++i)
#pragma unroll
        for (int j = 0; j < 4; ++j) acc[i][j] = wmma_bf16(a[i], b[j], acc[i][j]);
    }
  }

  const float scale = 0.17677669529663687f;  // hd^-0.5, folded into q
#pragma unroll
  for (int i = 0; i < 2; ++i) {
#pragma unroll
    for (int j = 0; j < 4; ++j) {
      int n = bn + wn + j * 16 + l16;        // column in [0,1536)
      int t = n >> 9;                        // 0=q 1=k 2=v
      int h = (n >> 5) & 15;
      int d = n & 31;
      float bv = bias[n];
      float sc = (t == 0) ? scale : 1.0f;
#pragma unroll
      for (int r = 0; r < 8; ++r) {
        unsigned m = (unsigned)(bm + wm + i * 16 + half8 + r);
        unsigned bw = m / 49u;
        unsigned s = m - bw * 49u;
        float v = (acc[i][j][r] + bv) * sc;
        qkv[(((size_t)t * BWIN + bw) * HEADS + h) * TILE_ELEMS + s * HD + d] = (__bf16)v;
      }
    }
  }
}

// ---------------- kernel 2: per-(window,head) attention, one wave each --------
// grid 8192, block 128 (4 waves)
__global__ __launch_bounds__(128) void attn_kernel(const __bf16* __restrict__ qkv,
                                                   const float* __restrict__ addtab,
                                                   __bf16* __restrict__ attnout) {
  __shared__ __bf16 Psh[4][SPAD * 72];   // softmax probs, row-major [m][k]
  __shared__ __bf16 Vt[4][HD * 72];      // v transposed [d][s]
  const int lane = threadIdx.x & 31, wave = threadIdx.x >> 5;
  const int l16 = lane & 15;
  const int kb0 = (lane >> 4) * 8;
  const int half8 = (lane >> 4) * 8;
  const int gw = blockIdx.x * 4 + wave;  // (b,h)
  const int b = gw >> 4, h = gw & 15;
  const int w = b & (NWIN - 1);

  const __bf16* qp = qkv + ((size_t)(0 * BWIN + b) * HEADS + h) * TILE_ELEMS;
  const __bf16* kp = qkv + ((size_t)(1 * BWIN + b) * HEADS + h) * TILE_ELEMS;
  const __bf16* vp = qkv + ((size_t)(2 * BWIN + b) * HEADS + h) * TILE_ELEMS;

  // ---- logits = q @ k^T (q pre-scaled) : 4x4 tiles, K=32 ----
  v16bf aq[4], bk[4];
#pragma unroll
  for (int mt = 0; mt < 4; ++mt) aq[mt] = load_frag(qp, mt * 16 + l16, HD, kb0);
#pragma unroll
  for (int nt = 0; nt < 4; ++nt) bk[nt] = load_frag(kp, nt * 16 + l16, HD, kb0);
  v8f c[4][4];
#pragma unroll
  for (int mt = 0; mt < 4; ++mt)
#pragma unroll
    for (int nt = 0; nt < 4; ++nt) c[mt][nt] = wmma_bf16(aq[mt], bk[nt], (v8f){});

  // ---- + (bias + mask) table, -1e30 in padding ----
  const float* at = addtab + ((size_t)(w * HEADS + h)) * (SPAD * SPAD);
#pragma unroll
  for (int mt = 0; mt < 4; ++mt)
#pragma unroll
    for (int nt = 0; nt < 4; ++nt)
#pragma unroll
      for (int r = 0; r < 8; ++r)
        c[mt][nt][r] += at[(mt * 16 + half8 + r) * SPAD + nt * 16 + l16];

  // ---- row softmax; rows of a 16x16 C tile live within one 16-lane half ----
  __bf16* Pw = &Psh[wave][0];
#pragma unroll
  for (int mt = 0; mt < 4; ++mt) {
    float rinv[8];
#pragma unroll
    for (int r = 0; r < 8; ++r) {
      float mx = fmaxf(fmaxf(c[mt][0][r], c[mt][1][r]),
                       fmaxf(c[mt][2][r], c[mt][3][r]));
#pragma unroll
      for (int xm = 1; xm < 16; xm <<= 1) mx = fmaxf(mx, __shfl_xor(mx, xm, 32));
      float s = 0.f;
#pragma unroll
      for (int nt = 0; nt < 4; ++nt) {
        float p = __expf(c[mt][nt][r] - mx);
        c[mt][nt][r] = p;
        s += p;
      }
#pragma unroll
      for (int xm = 1; xm < 16; xm <<= 1) s += __shfl_xor(s, xm, 32);
      rinv[r] = 1.0f / s;
    }
#pragma unroll
    for (int nt = 0; nt < 4; ++nt)
#pragma unroll
      for (int r = 0; r < 8; ++r)
        Pw[(mt * 16 + half8 + r) * 72 + nt * 16 + l16] =
            (__bf16)(c[mt][nt][r] * rinv[r]);
  }

  // ---- stage v transposed into LDS: Vt[d][s] ----
  __bf16* vt = &Vt[wave][0];
#pragma unroll 8
  for (int j = 0; j < SPAD; ++j) vt[lane * 72 + j] = vp[j * HD + lane];

  __syncthreads();

  // ---- out = P @ v : 4x2 tiles, K = 64 (2 steps) ----
  v8f o[4][2];
#pragma unroll
  for (int mt = 0; mt < 4; ++mt)
#pragma unroll
    for (int nt = 0; nt < 2; ++nt) o[mt][nt] = (v8f){};
#pragma unroll
  for (int kt = 0; kt < 2; ++kt) {
    v16bf ap[4], bv[2];
#pragma unroll
    for (int mt = 0; mt < 4; ++mt)
      ap[mt] = load_frag(Pw, mt * 16 + l16, 72, kt * 32 + kb0);
#pragma unroll
    for (int nt = 0; nt < 2; ++nt)
      bv[nt] = load_frag(vt, nt * 16 + l16, 72, kt * 32 + kb0);
#pragma unroll
    for (int mt = 0; mt < 4; ++mt)
#pragma unroll
      for (int nt = 0; nt < 2; ++nt) o[mt][nt] = wmma_bf16(ap[mt], bv[nt], o[mt][nt]);
  }

  // ---- store valid rows to attnout[b][s][h*32+d] (bf16) ----
#pragma unroll
  for (int mt = 0; mt < 4; ++mt)
#pragma unroll
    for (int nt = 0; nt < 2; ++nt)
#pragma unroll
      for (int r = 0; r < 8; ++r) {
        int s = mt * 16 + half8 + r;
        if (s < SWIN) {
          int d = nt * 16 + l16;
          attnout[((size_t)b * SWIN + s) * DIMC + h * HD + d] = (__bf16)o[mt][nt][r];
        }
      }
}

// ---------------- kernel 3: out = attnout @ proj_w + proj_b (f32 out) ---------
// grid (784, 4), block 256
__global__ __launch_bounds__(256) void proj_gemm(const __bf16* __restrict__ A,
                                                 const float* __restrict__ w,
                                                 const float* __restrict__ bias,
                                                 float* __restrict__ out) {
  __shared__ __bf16 As[128 * 72];
  __shared__ __bf16 Bs[128 * 72];
  const int tid = threadIdx.x;
  const int lane = tid & 31, wave = tid >> 5;
  const int l16 = lane & 15;
  const int kb0 = (lane >> 4) * 8;
  const int half8 = (lane >> 4) * 8;
  const int bm = blockIdx.x * 128;
  const int bn = blockIdx.y * 128;
  const int wm = (wave >> 1) * 32, wn = (wave & 1) * 64;

  v8f acc[2][4];
#pragma unroll
  for (int i = 0; i < 2; ++i)
#pragma unroll
    for (int j = 0; j < 4; ++j) acc[i][j] = (v8f){};

  for (int k0 = 0; k0 < DIMC; k0 += 64) {
    __syncthreads();
    {
      int r = tid >> 3;                 // A tile: 128 x 64 bf16 (already bf16)
      int cc = (tid & 7) * 8;
#pragma unroll
      for (int p = 0; p < 4; ++p, r += 32)
        *(v8bf*)(&As[r * 72 + cc]) =
            *(const v8bf*)(A + (size_t)(bm + r) * DIMC + k0 + cc);
      int kr = tid >> 5;                // B tile: 64 x 128 f32 -> transposed bf16
      int nc = (tid & 31) * 4;
#pragma unroll
      for (int p = 0; p < 8; ++p, kr += 8) {
        float4 f = *(const float4*)(w + (size_t)(k0 + kr) * DIMC + bn + nc);
        Bs[(nc + 0) * 72 + kr] = (__bf16)f.x;
        Bs[(nc + 1) * 72 + kr] = (__bf16)f.y;
        Bs[(nc + 2) * 72 + kr] = (__bf16)f.z;
        Bs[(nc + 3) * 72 + kr] = (__bf16)f.w;
      }
    }
    __syncthreads();
#pragma unroll
    for (int ks = 0; ks < 64; ks += 32) {
      v16bf a[2], b[4];
#pragma unroll
      for (int i = 0; i < 2; ++i) a[i] = load_frag(As, wm + i * 16 + l16, 72, ks + kb0);
#pragma unroll
      for (int j = 0; j < 4; ++j) b[j] = load_frag(Bs, wn + j * 16 + l16, 72, ks + kb0);
#pragma unroll
      for (int i = 0; i < 2; ++i)
#pragma unroll
        for (int j = 0; j < 4; ++j) acc[i][j] = wmma_bf16(a[i], b[j], acc[i][j]);
    }
  }

#pragma unroll
  for (int i = 0; i < 2; ++i)
#pragma unroll
    for (int j = 0; j < 4; ++j) {
      int n = bn + wn + j * 16 + l16;
      float bv = bias[n];
#pragma unroll
      for (int r = 0; r < 8; ++r) {
        int m = bm + wm + i * 16 + half8 + r;
        out[(size_t)m * DIMC + n] = acc[i][j][r] + bv;
      }
    }
}

// ---------------- host launch ----------------
extern "C" void kernel_launch(void* const* d_in, const int* in_sizes, int n_in,
                              void* d_out, int out_size, void* d_ws, size_t ws_size,
                              hipStream_t stream) {
  (void)in_sizes; (void)n_in; (void)out_size; (void)ws_size;
  const float* x          = (const float*)d_in[0];
  const float* mask       = (const float*)d_in[1];
  const float* qkv_w      = (const float*)d_in[2];
  const float* qkv_b      = (const float*)d_in[3];
  const float* proj_w     = (const float*)d_in[4];
  const float* proj_b     = (const float*)d_in[5];
  const float* bias_table = (const float*)d_in[6];
  const int*   rel_idx    = (const int*)d_in[7];

  char* ws = (char*)d_ws;
  float*  addtab  = (float*)ws;
  __bf16* qkv     = (__bf16*)(ws + QKV_OFF);
  __bf16* attnout = (__bf16*)(ws + ATT_OFF);

  build_add<<<(NWIN * HEADS * SPAD * SPAD) / 256, 256, 0, stream>>>(
      bias_table, rel_idx, mask, addtab);
  zero_pads<<<(3 * BWIN * HEADS * 60) / 256, 256, 0, stream>>>(qkv);
  qkv_gemm<<<dim3(MROWS / 128, NQKV / 128), 256, 0, stream>>>(x, qkv_w, qkv_b, qkv);
  attn_kernel<<<(BWIN * HEADS) / 4, 128, 0, stream>>>(qkv, addtab, attnout);
  proj_gemm<<<dim3(MROWS / 128, DIMC / 128), 256, 0, stream>>>(
      attnout, proj_w, proj_b, (float*)d_out);
}
